// GATrEncoder_27238682591845
// MI455X (gfx1250) — matrix operand order repaired and checked by reading, not compiled, not run on hardware
//
#include <hip/hip_runtime.h>
#include <math.h>

typedef __attribute__((ext_vector_type(16))) _Float16 v16h;
typedef __attribute__((ext_vector_type(8)))  _Float16 v8h;
typedef __attribute__((ext_vector_type(8)))  float    v8f;

#define NANCH 1022
#define NBINS 65536

// compile-time tables: folded into immediates after unrolling
constexpr int c_MASKS[16]  = {0,1,2,4,8,3,5,9,6,10,12,7,11,13,14,15};
constexpr int c_IDXOF[16]  = {0,1,2,5,3,6,8,11,4,7,9,12,10,13,14,15};
constexpr int c_GRADE[16]  = {0,1,1,1,1,2,2,2,2,2,2,3,3,3,3,4};
constexpr int c_OFFSRC[16] = {-1,0,-1,-1,-1,2,3,4,-1,-1,-1,8,9,10,-1,14};
constexpr int c_OFFB[16]   = { 0,5, 0, 0, 0,6,6,6, 0, 0, 0,7,7, 7, 0, 8};
constexpr int c_EU[8]      = {0,2,3,4,8,9,10,14};

__device__ __forceinline__ float swsign(int a, int b){
  int t = 0; a >>= 1;
  while (a){ t += __popc(a & b); a >>= 1; }
  return (t & 1) ? -1.f : 1.f;
}
__device__ __forceinline__ float gelu_t(float x){
  const float k0 = 0.7978845608028654f, k1 = 0.044715f;
  return 0.5f*x*(1.f + tanhf(k0*(x + k1*x*x*x)));
}
__device__ __forceinline__ int binof(float x){
  int b = (int)(x * 65536.0f);
  return b < 0 ? 0 : (b > 65535 ? 65535 : b);
}
__device__ __forceinline__ void embed_pt(float* dst, float px, float py, float pz){
  #pragma unroll
  for (int k = 0; k < 16; ++k) dst[k] = 0.f;
  dst[14] = 1.f; dst[13] = px; dst[12] = -py; dst[11] = pz;
}
__device__ __forceinline__ v16h join16(v8h lo, v8h hi){
  return __builtin_shufflevector(lo, hi, 0,1,2,3,4,5,6,7,8,9,10,11,12,13,14,15);
}

// ---------------- top-k selection ----------------
__global__ void k_hist(const float* __restrict__ op, int n, unsigned* __restrict__ hist){
  int stride = gridDim.x*blockDim.x;
  for (int i = blockIdx.x*blockDim.x + threadIdx.x; i < n; i += stride){
    __builtin_prefetch(op + i + stride, 0, 1);
    atomicAdd(&hist[binof(op[i])], 1u);
  }
}
__global__ void k_scan(const unsigned* __restrict__ hist, int* __restrict__ info){
  if (threadIdx.x == 0 && blockIdx.x == 0){
    unsigned acc = 0; int T = 0;
    for (int b = 65535; b >= 0; --b){
      unsigned c = hist[b];
      if (acc + c >= (unsigned)NANCH){ T = b; break; }
      acc += c;
    }
    info[0] = T; info[1] = NANCH - (int)acc;
  }
}
__global__ void k_special(const float* clspt, const float* tx, const float* temb,
                          float* GA, float* S0){
  if (threadIdx.x == 0 && blockIdx.x == 0){
    embed_pt(GA + 0,  clspt[0], clspt[1], clspt[2]);
    embed_pt(GA + 16, tx[0], tx[1], tx[2]);
    S0[0] = temb[0]; S0[1] = temb[1];
    S0[2] = temb[2]; S0[3] = temb[3];
  }
}
__global__ void k_select(const float* __restrict__ op, const float* __restrict__ pts, int n,
                         const int* __restrict__ info, int* __restrict__ cnt,
                         const float* __restrict__ temb, float* __restrict__ GA, float* __restrict__ S0){
  int T = info[0], need = info[1];
  int stride = gridDim.x*blockDim.x;
  for (int i = blockIdx.x*blockDim.x + threadIdx.x; i < n; i += stride){
    __builtin_prefetch(op + i + stride, 0, 1);
    int b = binof(op[i]);
    int slot = -1;
    if (b > T) slot = atomicAdd(&cnt[0], 1);
    else if (b == T){
      int q = atomicAdd(&cnt[1], 1);
      if (q < need) slot = (NANCH - need) + q;
    }
    if (slot >= 0 && slot < NANCH){
      int tok = 2 + slot;
      embed_pt(GA + (size_t)tok*16, pts[3*(size_t)i], pts[3*(size_t)i+1], pts[3*(size_t)i+2]);
      S0[tok*2+0] = temb[4]; S0[tok*2+1] = temb[5];
    }
  }
}

// ---------------- weight packing: wm(co,ci,9) -> WP[b][n][k] f16 ----------------
__global__ void k_packw(const float* __restrict__ wm, _Float16* __restrict__ WP){
  int id = blockIdx.x*blockDim.x + threadIdx.x;
  if (id >= 9*64*64) return;
  int k = id & 63, n = (id >> 6) & 63, b = id >> 12;
  WP[id] = (_Float16)wm[((size_t)n*64 + k)*9 + b];
}

// ---------------- lin_in (ci=1, csi=2 -> co=64, cso=4) ----------------
__global__ void k_lin_in(const float* __restrict__ GA, const float* __restrict__ S0,
    const float* __restrict__ wm, const float* __restrict__ bm, const float* __restrict__ ws2m,
    const float* __restrict__ ws, const float* __restrict__ wm2s, const float* __restrict__ bs,
    float* __restrict__ X, float* __restrict__ XS){
  int t = blockIdx.x, o = threadIdx.x;
  __shared__ float ga[16], s0[2];
  if (o < 16) ga[o] = GA[t*16+o];
  if (o < 2)  s0[o] = S0[t*2+o];
  __syncthreads();
  #pragma unroll
  for (int j = 0; j < 16; ++j){
    float v = wm[o*9 + c_GRADE[j]] * ga[j];
    constexpr int srcs[16] = {-1,0,-1,-1,-1,2,3,4,-1,-1,-1,8,9,10,-1,14};
    int src = srcs[j];
    if (src >= 0) v += wm[o*9 + c_OFFB[j]] * ga[src];
    if (j == 0) v += ws2m[o*2+0]*s0[0] + ws2m[o*2+1]*s0[1] + bm[o];
    X[((size_t)(t*64+o))*16 + j] = v;
  }
  if (o < 4)
    XS[t*4+o] = bs[o] + ws[o*2+0]*s0[0] + ws[o*2+1]*s0[1] + wm2s[o]*ga[0];
}

// ---------------- equivariant linear, WMMA core (64ch -> 64ch) ----------------
// LDS holds A fragments: [comp*2+kstep][lane][e], 32B contiguous per lane.
__global__ __launch_bounds__(128) void k_el64(const float* __restrict__ in_mv,
        const _Float16* __restrict__ WP, float* __restrict__ out_mv){
  __shared__ _Float16 lds[32*32*16];
  int t0 = blockIdx.x * 16;
  for (int i = threadIdx.x; i < 16384; i += 128){
    int comp = i & 15, ch = (i >> 4) & 63, m = i >> 10;
    float val = in_mv[(((size_t)(t0+m))*64 + ch)*16 + comp];
    int ks = ch >> 5, r = ch & 31;
    int gA2 = r >> 4;
    int e   = r & 15;
    int lane2 = (gA2 << 4) + m;
    lds[(((comp << 1) + ks)*32 + lane2)*16 + e] = (_Float16)val;
  }
  __syncthreads();
  int lane = threadIdx.x & 31, wave = threadIdx.x >> 5;
  int n0 = wave * 16;
  int nB = lane & 15, hB = lane >> 4;
  #pragma unroll
  for (int j = 0; j < 16; ++j){
    v8f acc = {};
    const int bidx = c_GRADE[j];
    #pragma unroll
    for (int ks = 0; ks < 2; ++ks){
      v16h a = *(const v16h*)(lds + (((j << 1) + ks)*32 + lane)*16);
      v16h b = *(const v16h*)(WP + ((size_t)(bidx*64 + n0 + nB))*64 + ks*32 + hB*16);
      acc = __builtin_amdgcn_wmma_f32_16x16x32_f16(false, a, false, b, (short)0, acc, false, false);
    }
    const int src = c_OFFSRC[j];
    if (src >= 0){
      const int b2 = c_OFFB[j];
      #pragma unroll
      for (int ks = 0; ks < 2; ++ks){
        v16h a = *(const v16h*)(lds + (((src << 1) + ks)*32 + lane)*16);
        v16h b = *(const v16h*)(WP + ((size_t)(b2*64 + n0 + nB))*64 + ks*32 + hB*16);
        acc = __builtin_amdgcn_wmma_f32_16x16x32_f16(false, a, false, b, (short)0, acc, false, false);
      }
    }
    #pragma unroll
    for (int v = 0; v < 8; ++v){
      int m = v + ((lane >> 4) << 3);
      int n = lane & 15;
      out_mv[(((size_t)(t0+m))*64 + (n0+n))*16 + j] = acc[v];
    }
  }
}

// ---------------- scalar channel path of el (csi=cso=4) ----------------
__global__ void k_el_scalar(const float* __restrict__ in_mv, const float* __restrict__ in_s,
    const float* __restrict__ ws2m, const float* __restrict__ bm, const float* __restrict__ ws,
    const float* __restrict__ wm2s, const float* __restrict__ bs,
    float* __restrict__ out_mv, float* __restrict__ out_s){
  int t = blockIdx.x, o = threadIdx.x;
  __shared__ float sl[4];
  if (o < 4) sl[o] = in_s[t*4+o];
  __syncthreads();
  float add = bm[o];
  #pragma unroll
  for (int k2 = 0; k2 < 4; ++k2) add += ws2m[o*4+k2]*sl[k2];
  out_mv[((size_t)(t*64+o))*16 + 0] += add;
  if (o < 4){
    float v = bs[o];
    #pragma unroll
    for (int k2 = 0; k2 < 4; ++k2) v += ws[o*4+k2]*sl[k2];
    for (int i = 0; i < 64; ++i) v += wm2s[o*64+i]*in_mv[((size_t)(t*64+i))*16 + 0];
    out_s[t*4+o] = v;
  }
}

// ---------------- LayerNorm-like equinorm ----------------
__global__ void k_ln(const float* __restrict__ in_mv, const float* __restrict__ in_s,
                     float* __restrict__ out_mv, float* __restrict__ out_s){
  int t = blockIdx.x, c = threadIdx.x;
  __shared__ float red[64];
  float v[16];
  #pragma unroll
  for (int j = 0; j < 16; ++j) v[j] = in_mv[((size_t)(t*64+c))*16 + j];
  float sq = 0.f;
  #pragma unroll
  for (int e = 0; e < 8; ++e){ float x = v[c_EU[e]]; sq += x*x; }
  red[c] = sq; __syncthreads();
  for (int st = 32; st > 0; st >>= 1){ if (c < st) red[c] += red[c+st]; __syncthreads(); }
  float inv = rsqrtf(red[0]*(1.f/64.f) + 1e-6f);
  #pragma unroll
  for (int j = 0; j < 16; ++j) out_mv[((size_t)(t*64+c))*16 + j] = v[j]*inv;
  if (c < 4){
    float ss = 0.f;
    #pragma unroll
    for (int k2 = 0; k2 < 4; ++k2){ float x = in_s[t*4+k2]; ss += x*x; }
    out_s[t*4+c] = in_s[t*4+c]*rsqrtf(ss*0.25f + 1e-6f);
  }
}

// ---------------- attention ----------------
__global__ void k_pack(const float* __restrict__ mv, _Float16* __restrict__ F){
  int id = blockIdx.x*blockDim.x + threadIdx.x;
  if (id >= 4*1024*128) return;
  int f = id & 127, t = (id >> 7) & 1023, h = id >> 17;
  int c = f & 15, e = f >> 4;
  F[id] = (_Float16)mv[(((size_t)t)*64 + h*16 + c)*16 + c_EU[e]];
}
// V transposed per head: VFT[h][g][u], g = c*16+comp (256 per head)
__global__ void k_packvt(const float* __restrict__ mv, _Float16* __restrict__ VFT){
  int id = blockIdx.x*blockDim.x + threadIdx.x;
  if (id >= 4*256*1024) return;
  int u = id & 1023, g = (id >> 10) & 255, h = id >> 18;
  int c = g >> 4, comp = g & 15;
  VFT[id] = (_Float16)mv[(((size_t)u)*64 + h*16 + c)*16 + comp];
}
__global__ void k_logits(const _Float16* __restrict__ QF, const _Float16* __restrict__ KF,
                         const float* __restrict__ qsv, const float* __restrict__ ksv,
                         float* __restrict__ LOGB){
  int mt = blockIdx.x, nt = blockIdx.y, h = blockIdx.z, lane = threadIdx.x;
  int t0 = mt*16, u0 = nt*16;
  const _Float16* Qh = QF + (size_t)h*1024*128;
  const _Float16* Kh = KF + (size_t)h*1024*128;
  int mA = lane & 15, gA = lane >> 4, nB = lane & 15, hB = lane >> 4;
  const _Float16* arow = Qh + (size_t)(t0+mA)*128 + gA*16;
  const _Float16* brow = Kh + (size_t)(u0+nB)*128 + hB*16;
  v8f acc = {};
  #pragma unroll
  for (int ks = 0; ks < 4; ++ks){
    int kb = ks*32;
    v16h a = *(const v16h*)(arow + kb);
    v16h b = *(const v16h*)(brow + kb);
    acc = __builtin_amdgcn_wmma_f32_16x16x32_f16(false, a, false, b, (short)0, acc, false, false);
  }
  const float scale = 0.08804509063256238f; // 1/sqrt(8*16+1)
  #pragma unroll
  for (int v = 0; v < 8; ++v){
    int t = t0 + v + ((lane >> 4) << 3);
    int u = u0 + (lane & 15);
    float qs = qsv[t*4+h], ksx = ksv[u*4+h];
    LOGB[((size_t)h*1024 + t)*1024 + u] = (acc[v] + qs*ksx)*scale;
  }
}
__global__ void k_softmax(const float* __restrict__ LOGB, _Float16* __restrict__ WF){
  int t = blockIdx.x, h = blockIdx.y, lane = threadIdx.x;
  const float* row = LOGB + ((size_t)h*1024 + t)*1024;
  float mx = -1e30f;
  for (int u = lane; u < 1024; u += 32) mx = fmaxf(mx, row[u]);
  for (int o2 = 16; o2 > 0; o2 >>= 1) mx = fmaxf(mx, __shfl_xor(mx, o2));
  float sum = 0.f;
  for (int u = lane; u < 1024; u += 32) sum += __expf(row[u]-mx);
  for (int o2 = 16; o2 > 0; o2 >>= 1) sum += __shfl_xor(sum, o2);
  float inv = 1.f/sum;
  _Float16* wrow = WF + ((size_t)h*1024 + t)*1024;
  for (int u = lane; u < 1024; u += 32) wrow[u] = (_Float16)(__expf(row[u]-mx)*inv);
}
__global__ void k_om(const _Float16* __restrict__ WF, const _Float16* __restrict__ VFT,
                     float* __restrict__ OM){
  int mt = blockIdx.x, gt = blockIdx.y, h = blockIdx.z, lane = threadIdx.x;
  int t0 = mt*16, g0 = gt*16;
  int mA = lane & 15, gA = lane >> 4, nB = lane & 15, hB = lane >> 4;
  const _Float16* arow = WF + ((size_t)h*1024 + (t0+mA))*1024 + gA*16;
  const _Float16* brow = VFT + ((size_t)h*256 + (g0+nB))*1024 + hB*16;
  v8f acc = {};
  for (int ks = 0; ks < 32; ++ks){
    int kb = ks*32;
    v16h a = *(const v16h*)(arow + kb);
    v16h b = *(const v16h*)(brow + kb);
    acc = __builtin_amdgcn_wmma_f32_16x16x32_f16(false, a, false, b, (short)0, acc, false, false);
  }
  #pragma unroll
  for (int v = 0; v < 8; ++v){
    int t = t0 + v + ((lane >> 4) << 3);
    int g = g0 + (lane & 15);
    OM[((size_t)t*64 + h*16 + (g>>4))*16 + (g&15)] = acc[v];
  }
}
__global__ void k_os(const _Float16* __restrict__ WF, const float* __restrict__ VS,
                     float* __restrict__ OS){
  int t = blockIdx.x, h = blockIdx.y, lane = threadIdx.x;
  const _Float16* wrow = WF + ((size_t)h*1024 + t)*1024;
  float s = 0.f;
  for (int u = lane; u < 1024; u += 32) s += (float)wrow[u]*VS[u*4+h];
  for (int o2 = 16; o2 > 0; o2 >>= 1) s += __shfl_xor(s, o2);
  if (lane == 0) OS[t*4+h] = s;
}
__global__ void k_add(float* __restrict__ dst, const float* __restrict__ src, int n){
  int i = blockIdx.x*blockDim.x + threadIdx.x;
  if (i < n) dst[i] += src[i];
}

// ---------------- geometric product / join + gating ----------------
__global__ void k_bilinear(const float* __restrict__ L, const float* __restrict__ Ls,
                           const float* __restrict__ R, const float* __restrict__ Rs,
                           float* __restrict__ HID, float* __restrict__ Hs){
  int id = blockIdx.x*blockDim.x + threadIdx.x;
  if (id >= 1024*64) return;
  int t = id >> 6, c = id & 63;
  float l[16], r[16], o[16];
  #pragma unroll
  for (int k = 0; k < 16; ++k){ l[k] = L[(size_t)id*16+k]; r[k] = R[(size_t)id*16+k]; o[k] = 0.f; }
  if (c < 32){
    #pragma unroll
    for (int i = 0; i < 16; ++i){
      const int a = c_MASKS[i];
      #pragma unroll
      for (int j = 0; j < 16; ++j){
        const int b = c_MASKS[j];
        if (a & b & 1) continue;
        o[c_IDXOF[a^b]] += swsign(a,b)*l[i]*r[j];
      }
    }
  } else {
    float da[16], db[16], w[16];
    #pragma unroll
    for (int k = 0; k < 16; ++k){
      const int i = c_IDXOF[15 ^ c_MASKS[k]];
      float sg = swsign(c_MASKS[i], c_MASKS[k]);
      da[k] = sg*l[i]; db[k] = sg*r[i]; w[k] = 0.f;
    }
    #pragma unroll
    for (int i = 0; i < 16; ++i){
      const int a = c_MASKS[i];
      #pragma unroll
      for (int j = 0; j < 16; ++j){
        const int b = c_MASKS[j];
        if (a & b) continue;
        w[c_IDXOF[a^b]] += swsign(a,b)*da[i]*db[j];
      }
    }
    #pragma unroll
    for (int k = 0; k < 16; ++k){
      const int i = c_IDXOF[15 ^ c_MASKS[k]];
      o[k] = swsign(c_MASKS[i], c_MASKS[k]) * w[i];
    }
  }
  float g = gelu_t(o[0]);
  #pragma unroll
  for (int k = 0; k < 16; ++k) HID[(size_t)id*16+k] = o[k]*g;
  if (c < 4) Hs[t*4+c] = gelu_t(Ls[t*4+c] + Rs[t*4+c]);
}

// ---------------- lin_out (token 0 only) + cls projection ----------------
__global__ void k_final(const float* __restrict__ X, const float* __restrict__ XS,
    const float* __restrict__ wm, const float* __restrict__ bm, const float* __restrict__ ws2m,
    const float* __restrict__ ws, const float* __restrict__ wm2s, const float* __restrict__ bs,
    const float* __restrict__ pw, const float* __restrict__ pb, float* __restrict__ out){
  __shared__ float cls[18];
  int tid = threadIdx.x;
  if (tid < 16){
    int j = tid;
    float v = 0.f;
    int b = c_GRADE[j];
    for (int i = 0; i < 64; ++i) v += wm[i*9+b]*X[i*16+j];
    int src = c_OFFSRC[j];
    if (src >= 0){ int b2 = c_OFFB[j]; for (int i = 0; i < 64; ++i) v += wm[i*9+b2]*X[i*16+src]; }
    if (j == 0){ for (int k2 = 0; k2 < 4; ++k2) v += ws2m[k2]*XS[k2]; v += bm[0]; }
    cls[j] = v;
  } else if (tid < 18){
    int c2 = tid - 16;
    float v = bs[c2];
    for (int k2 = 0; k2 < 4; ++k2) v += ws[c2*4+k2]*XS[k2];
    for (int i = 0; i < 64; ++i) v += wm2s[c2*64+i]*X[i*16+0];
    cls[tid] = v;
  }
  __syncthreads();
  float r = pb[tid];
  for (int f = 0; f < 18; ++f) r += pw[tid*18+f]*cls[f];
  out[tid] = r;
}

extern "C" void kernel_launch(void* const* d_in, const int* in_sizes, int n_in,
                              void* d_out, int out_size, void* d_ws, size_t ws_size,
                              hipStream_t stream) {
  (void)out_size;
  if (n_in < 62) return;
  const float* pts   = (const float*)d_in[0];
  const float* tx    = (const float*)d_in[2];
  const float* opac  = (const float*)d_in[3];
  const float* clspt = (const float*)d_in[4];
  const float* temb  = (const float*)d_in[5];
  int N = in_sizes[3];

  char* W = (char*)d_ws;
  size_t off = 0;
  auto alloc = [&](size_t b){ size_t o = off; off += (b + 255) & ~(size_t)255; return o; };
  unsigned* HIST = (unsigned*)(W + alloc((size_t)NBINS*4));
  int* INFO = (int*)(W + alloc(256));
  int* CNT  = (int*)(W + alloc(256));
  float* GA = (float*)(W + alloc((size_t)1024*16*4));
  float* S0 = (float*)(W + alloc((size_t)1024*2*4));
  size_t MVB = (size_t)1024*64*16*4, SCB = (size_t)1024*4*4;
  float* X   = (float*)(W + alloc(MVB)); float* XS  = (float*)(W + alloc(SCB));
  float* H   = (float*)(W + alloc(MVB)); float* HSb = (float*)(W + alloc(SCB));
  float* QM  = (float*)(W + alloc(MVB)); float* QS  = (float*)(W + alloc(SCB));
  float* KM  = (float*)(W + alloc(MVB)); float* KS  = (float*)(W + alloc(SCB));
  float* VM  = (float*)(W + alloc(MVB)); float* VS  = (float*)(W + alloc(SCB));
  float* OM  = (float*)(W + alloc(MVB)); float* OS  = (float*)(W + alloc(SCB));
  float* T1  = (float*)(W + alloc(MVB)); float* T1S = (float*)(W + alloc(SCB));
  _Float16* QF  = (_Float16*)(W + alloc((size_t)4*1024*128*2));
  _Float16* KF  = (_Float16*)(W + alloc((size_t)4*1024*128*2));
  _Float16* VFT = (_Float16*)(W + alloc((size_t)4*256*1024*2));
  float*    LG  = (float*)   (W + alloc((size_t)4*1024*1024*4));
  _Float16* WF  = (_Float16*)(W + alloc((size_t)4*1024*1024*2));
  // packed f16 weights, one per el64 layer (q,k,v,o,left,right,down)
  const int elbase[7] = {12, 18, 24, 30, 36, 42, 48};
  _Float16* WP[7];
  for (int w7 = 0; w7 < 7; ++w7) WP[w7] = (_Float16*)(W + alloc((size_t)9*64*64*2));
  if (off > ws_size) return;

  auto launch_el = [&](const float* in_mv, const float* in_s, int li, float* out_mv, float* out_s){
    int base = elbase[li];
    k_el64<<<64, 128, 0, stream>>>(in_mv, WP[li], out_mv);
    k_el_scalar<<<1024, 64, 0, stream>>>(in_mv, in_s,
        (const float*)d_in[base+2], (const float*)d_in[base+1],
        (const float*)d_in[base+3], (const float*)d_in[base+4], (const float*)d_in[base+5],
        out_mv, out_s);
  };

  // pack weights (tiny, L2 resident afterwards)
  for (int w7 = 0; w7 < 7; ++w7)
    k_packw<<<144, 256, 0, stream>>>((const float*)d_in[elbase[w7]], WP[w7]);

  // top-k select + embed
  hipMemsetAsync(HIST, 0, (size_t)NBINS*4, stream);
  hipMemsetAsync(CNT,  0, 256, stream);
  int gsel = (N + 255)/256; if (gsel > 4096) gsel = 4096;
  k_hist<<<gsel, 256, 0, stream>>>(opac, N, HIST);
  k_scan<<<1, 32, 0, stream>>>(HIST, INFO);
  k_special<<<1, 32, 0, stream>>>(clspt, tx, temb, GA, S0);
  k_select<<<gsel, 256, 0, stream>>>(opac, pts, N, INFO, CNT, temb, GA, S0);

  // lin_in
  k_lin_in<<<1024, 64, 0, stream>>>(GA, S0,
      (const float*)d_in[6], (const float*)d_in[7], (const float*)d_in[8],
      (const float*)d_in[9], (const float*)d_in[10], (const float*)d_in[11], X, XS);

  // ---- block: attention ----
  k_ln<<<1024, 64, 0, stream>>>(X, XS, H, HSb);
  launch_el(H, HSb, 0, QM, QS);
  launch_el(H, HSb, 1, KM, KS);
  launch_el(H, HSb, 2, VM, VS);
  k_pack<<<2048, 256, 0, stream>>>(QM, QF);
  k_pack<<<2048, 256, 0, stream>>>(KM, KF);
  k_packvt<<<4096, 256, 0, stream>>>(VM, VFT);
  k_logits<<<dim3(64,64,4), 32, 0, stream>>>(QF, KF, QS, KS, LG);
  k_softmax<<<dim3(1024,4), 32, 0, stream>>>(LG, WF);
  k_om<<<dim3(64,16,4), 32, 0, stream>>>(WF, VFT, OM);
  k_os<<<dim3(1024,4), 32, 0, stream>>>(WF, VS, OS);
  launch_el(OM, OS, 3, T1, T1S);
  k_add<<<4096, 256, 0, stream>>>(X, T1, 1024*64*16);
  k_add<<<16, 256, 0, stream>>>(XS, T1S, 1024*4);

  // ---- block: geometric MLP ----
  k_ln<<<1024, 64, 0, stream>>>(X, XS, H, HSb);
  launch_el(H, HSb, 4, QM, QS);
  launch_el(H, HSb, 5, KM, KS);
  k_bilinear<<<256, 256, 0, stream>>>(QM, QS, KM, KS, VM, VS);
  launch_el(VM, VS, 6, T1, T1S);
  k_add<<<4096, 256, 0, stream>>>(X, T1, 1024*64*16);
  k_add<<<16, 256, 0, stream>>>(XS, T1S, 1024*4);

  // ---- lin_out + cls projection (token 0 only) ----
  k_final<<<1, 64, 0, stream>>>(X, XS,
      (const float*)d_in[54], (const float*)d_in[55], (const float*)d_in[56],
      (const float*)d_in[57], (const float*)d_in[58], (const float*)d_in[59],
      (const float*)d_in[60], (const float*)d_in[61], (float*)d_out);
}
